// SymbolicFlowMatching_54666343744035
// MI455X (gfx1250) — matrix operand order repaired and checked by reading, not compiled, not run on hardware
//
#include <hip/hip_runtime.h>

// Problem constants (B=16, L=256, D=512, V=32000)
#define DDIM     512
#define VROWS    32000
#define BLROWS   4096
#define KCHUNKS  16        // 512 / 32 (K per wmma)
#define NTILES   2000      // 32000 / 16
#define NWAVES   8
#define MT       2         // 16-row M tiles per block -> 32 rows/block
#define ROWS_PB  32
#define NB       4         // N-tiles per wave batch
#define VSLICES  2
#define SL_TILES (NTILES / VSLICES)   // 1000 N-tiles per slice
#define SA_STRIDE 520      // padded LDS row stride (ushorts) -> bank-conflict free

typedef __attribute__((ext_vector_type(16))) __bf16        v16bf;
typedef __attribute__((ext_vector_type(8)))  float         v8f;
typedef __attribute__((ext_vector_type(4)))  unsigned int  u32x4;

struct RawFrag { u32x4 lo, hi; };   // 32 bytes == sizeof(v16bf)

static __device__ __forceinline__ unsigned short f2bf(float f) {
    unsigned int u = __float_as_uint(f);
    u += 0x7FFFu + ((u >> 16) & 1u);          // round-to-nearest-even
    return (unsigned short)(u >> 16);
}

// ---- kernel 1: xt (fp32) -> bf16, 4 elems/thread -------------------------
__global__ void cvt_x_kernel(const float* __restrict__ x,
                             unsigned short* __restrict__ xb) {
    int i = (blockIdx.x * 256 + threadIdx.x) * 4;
    float4 f = *(const float4*)(x + i);
    uint2 o;
    o.x = (unsigned int)f2bf(f.x) | ((unsigned int)f2bf(f.y) << 16);
    o.y = (unsigned int)f2bf(f.z) | ((unsigned int)f2bf(f.w) << 16);
    *(uint2*)(xb + i) = o;
}

// ---- kernel 2: emb (fp32) -> bf16 + row sum-of-squares -------------------
__global__ void cvt_emb_kernel(const float* __restrict__ e,
                               unsigned short* __restrict__ eb,
                               float* __restrict__ esq) {
    int row  = blockIdx.x * 8 + (threadIdx.x >> 5);   // 8 rows / block
    int lane = threadIdx.x & 31;
    const float* er = e + (size_t)row * DDIM;
    unsigned short* ob = eb + (size_t)row * DDIM;
    float s = 0.0f;
#pragma unroll
    for (int j = 0; j < DDIM / 32; ++j) {
        float f = er[j * 32 + lane];
        s = __builtin_fmaf(f, f, s);
        ob[j * 32 + lane] = f2bf(f);
    }
#pragma unroll
    for (int off = 16; off >= 1; off >>= 1)
        s += __shfl_xor(s, off, 32);
    if (lane == 0) esq[row] = s;
}

// ---- kernel 3: WMMA GEMM + per-slice argmin ------------------------------
// grid = (BLROWS/32, VSLICES); block = 256 (8 waves). Each block: 32 rows x
// one V-slice. Register tile 2M x 4N: per K-chunk iteration the wave does
// 4 ds_load_b128 (A) + 8 global_load_b128 (B, 4 streams) + 8 WMMAs.
__global__ void __launch_bounds__(256, 1)
gemm_argmin_kernel(const unsigned short* __restrict__ xb,
                   const unsigned short* __restrict__ eb,
                   const float* __restrict__ esq,
                   float* __restrict__ cand_d2,
                   int* __restrict__ cand_idx) {
    const int tid  = threadIdx.x;
    const int wave = tid >> 5;
    const int lane = tid & 31;
    const int half = lane >> 4;
    const int l16  = lane & 15;
    const int rowBase  = blockIdx.x * ROWS_PB;
    const int tileBase = blockIdx.y * SL_TILES;

    // Stage the 32x512 bf16 x-tile in LDS with padded rows (bank-friendly).
    __shared__ unsigned short sA[ROWS_PB * SA_STRIDE];
    {
        const unsigned short* src = xb + (size_t)rowBase * DDIM;
#pragma unroll
        for (int i = 0; i < 8; ++i) {
            int e   = (i * 256 + tid) * 8;   // 8 ushorts = 16B per copy
            int row = e >> 9;
            int col = e & 511;
            *(u32x4*)(sA + row * SA_STRIDE + col) = *(const u32x4*)(src + e);
        }
    }
    __syncthreads();

    float bd2[MT][8];
    int   bidx[MT][8];
#pragma unroll
    for (int m = 0; m < MT; ++m)
#pragma unroll
        for (int r = 0; r < 8; ++r) { bd2[m][r] = 3.0e38f; bidx[m][r] = 0; }

    // Per-lane A-frag base addresses in LDS (row = l16 within each M-tile).
    const unsigned short* aBase0 = sA + l16 * SA_STRIDE + 16 * half;
    const unsigned short* aBase1 = aBase0 + 16 * SA_STRIDE;

    for (int t0 = tileBase + wave * NB; t0 < tileBase + SL_TILES;
         t0 += NWAVES * NB) {
        const int n0 = t0 * 16 + l16;
        const int n1 = n0 + 16;
        const int n2 = n0 + 32;
        const int n3 = n0 + 48;
        const unsigned short* e0 = eb + (size_t)n0 * DDIM + 16 * half;
        const unsigned short* e1 = e0 + 16 * DDIM;
        const unsigned short* e2 = e0 + 32 * DDIM;
        const unsigned short* e3 = e0 + 48 * DDIM;
        const unsigned short* pa0 = aBase0;
        const unsigned short* pa1 = aBase1;

        v8f c00 = {0.f,0.f,0.f,0.f,0.f,0.f,0.f,0.f};
        v8f c01 = c00, c02 = c00, c03 = c00;
        v8f c10 = c00, c11 = c00, c12 = c00, c13 = c00;

#pragma unroll 1   // real loop: keeps A-frag LDS loads inside (no 256-reg hoist)
        for (int c = 0; c < KCHUNKS; ++c) {
            RawFrag rb0, rb1, rb2, rb3, ra0, ra1;
            rb0.lo = *(const u32x4*)(e0);      rb0.hi = *(const u32x4*)(e0 + 8);
            rb1.lo = *(const u32x4*)(e1);      rb1.hi = *(const u32x4*)(e1 + 8);
            rb2.lo = *(const u32x4*)(e2);      rb2.hi = *(const u32x4*)(e2 + 8);
            rb3.lo = *(const u32x4*)(e3);      rb3.hi = *(const u32x4*)(e3 + 8);
            ra0.lo = *(const u32x4*)(pa0);     ra0.hi = *(const u32x4*)(pa0 + 8);
            ra1.lo = *(const u32x4*)(pa1);     ra1.hi = *(const u32x4*)(pa1 + 8);
            v16bf b0 = __builtin_bit_cast(v16bf, rb0);
            v16bf b1 = __builtin_bit_cast(v16bf, rb1);
            v16bf b2 = __builtin_bit_cast(v16bf, rb2);
            v16bf b3 = __builtin_bit_cast(v16bf, rb3);
            v16bf f0 = __builtin_bit_cast(v16bf, ra0);
            v16bf f1 = __builtin_bit_cast(v16bf, ra1);
            c00 = __builtin_amdgcn_wmma_f32_16x16x32_bf16(
                      false, f0, false, b0, (short)0, c00, false, false);
            c10 = __builtin_amdgcn_wmma_f32_16x16x32_bf16(
                      false, f1, false, b0, (short)0, c10, false, false);
            c01 = __builtin_amdgcn_wmma_f32_16x16x32_bf16(
                      false, f0, false, b1, (short)0, c01, false, false);
            c11 = __builtin_amdgcn_wmma_f32_16x16x32_bf16(
                      false, f1, false, b1, (short)0, c11, false, false);
            c02 = __builtin_amdgcn_wmma_f32_16x16x32_bf16(
                      false, f0, false, b2, (short)0, c02, false, false);
            c12 = __builtin_amdgcn_wmma_f32_16x16x32_bf16(
                      false, f1, false, b2, (short)0, c12, false, false);
            c03 = __builtin_amdgcn_wmma_f32_16x16x32_bf16(
                      false, f0, false, b3, (short)0, c03, false, false);
            c13 = __builtin_amdgcn_wmma_f32_16x16x32_bf16(
                      false, f1, false, b3, (short)0, c13, false, false);
            e0 += 32; e1 += 32; e2 += 32; e3 += 32;
            pa0 += 32; pa1 += 32;
        }

        const float es0 = esq[n0];
        const float es1 = esq[n1];
        const float es2 = esq[n2];
        const float es3 = esq[n3];
#pragma unroll
        for (int r = 0; r < 8; ++r) {
            float d;
            d = __builtin_fmaf(-2.0f, c00[r], es0);
            if (d < bd2[0][r]) { bd2[0][r] = d; bidx[0][r] = n0; }
            d = __builtin_fmaf(-2.0f, c01[r], es1);
            if (d < bd2[0][r]) { bd2[0][r] = d; bidx[0][r] = n1; }
            d = __builtin_fmaf(-2.0f, c02[r], es2);
            if (d < bd2[0][r]) { bd2[0][r] = d; bidx[0][r] = n2; }
            d = __builtin_fmaf(-2.0f, c03[r], es3);
            if (d < bd2[0][r]) { bd2[0][r] = d; bidx[0][r] = n3; }
            d = __builtin_fmaf(-2.0f, c10[r], es0);
            if (d < bd2[1][r]) { bd2[1][r] = d; bidx[1][r] = n0; }
            d = __builtin_fmaf(-2.0f, c11[r], es1);
            if (d < bd2[1][r]) { bd2[1][r] = d; bidx[1][r] = n1; }
            d = __builtin_fmaf(-2.0f, c12[r], es2);
            if (d < bd2[1][r]) { bd2[1][r] = d; bidx[1][r] = n2; }
            d = __builtin_fmaf(-2.0f, c13[r], es3);
            if (d < bd2[1][r]) { bd2[1][r] = d; bidx[1][r] = n3; }
        }
    }

    // Cross-lane min within each 16-lane group (lanes sharing a C-row).
#pragma unroll
    for (int m = 0; m < MT; ++m)
#pragma unroll
        for (int r = 0; r < 8; ++r)
#pragma unroll
            for (int off = 8; off >= 1; off >>= 1) {
                float od = __shfl_xor(bd2[m][r], off, 32);
                int   oi = __shfl_xor(bidx[m][r], off, 32);
                if (od < bd2[m][r] ||
                    (od == bd2[m][r] && oi < bidx[m][r])) {
                    bd2[m][r] = od; bidx[m][r] = oi;
                }
            }

    __shared__ float s_d2[NWAVES][ROWS_PB];
    __shared__ int   s_idx[NWAVES][ROWS_PB];
    if (l16 == 0) {
#pragma unroll
        for (int m = 0; m < MT; ++m)
#pragma unroll
            for (int r = 0; r < 8; ++r) {
                s_d2[wave][m * 16 + r + 8 * half]  = bd2[m][r];
                s_idx[wave][m * 16 + r + 8 * half] = bidx[m][r];
            }
    }
    __syncthreads();

    if (tid < ROWS_PB) {
        float best = s_d2[0][tid];
        int   bi   = s_idx[0][tid];
#pragma unroll
        for (int w = 1; w < NWAVES; ++w) {
            float d  = s_d2[w][tid];
            int   i2 = s_idx[w][tid];
            if (d < best || (d == best && i2 < bi)) { best = d; bi = i2; }
        }
        int g = blockIdx.y * BLROWS + rowBase + tid;
        cand_d2[g]  = best;
        cand_idx[g] = bi;
    }
}

// ---- kernel 4: combine V-slices + exact fp32 gather ----------------------
// grid = 64 blocks x 256 threads; 64 rows per block.
__global__ void combine_gather_kernel(const float* __restrict__ cand_d2,
                                      const int* __restrict__ cand_idx,
                                      const float* __restrict__ emb,
                                      float* __restrict__ out) {
    __shared__ int s_best[64];
    const int tid = threadIdx.x;
    const int rowBase = blockIdx.x * 64;
    if (tid < 64) {
        int g = rowBase + tid;
        float d0 = cand_d2[g];           int i0 = cand_idx[g];
        float d1 = cand_d2[BLROWS + g];  int i1 = cand_idx[BLROWS + g];
        s_best[tid] = (d1 < d0 || (d1 == d0 && i1 < i0)) ? i1 : i0;
    }
    __syncthreads();
    // 64 rows x 128 float4 = 8192 float4 per block / 256 threads = 32 each.
#pragma unroll 4
    for (int i = 0; i < 32; ++i) {
        int idx = i * 256 + tid;
        int row = idx >> 7;
        int c4  = idx & 127;
        float4 v = *(const float4*)(emb + (size_t)s_best[row] * DDIM + c4 * 4);
        *(float4*)(out + (size_t)(rowBase + row) * DDIM + c4 * 4) = v;
    }
}

// ---- launcher ------------------------------------------------------------
extern "C" void kernel_launch(void* const* d_in, const int* in_sizes, int n_in,
                              void* d_out, int out_size, void* d_ws, size_t ws_size,
                              hipStream_t stream) {
    const float* xt  = (const float*)d_in[0];   // [4096, 512] fp32
    const float* emb = (const float*)d_in[1];   // [32000, 512] fp32
    float* out = (float*)d_out;                 // [4096, 512] fp32

    // Workspace layout (~35.5 MB total)
    unsigned char* ws = (unsigned char*)d_ws;
    unsigned short* xb      = (unsigned short*)(ws);                  //  4,194,304 B
    unsigned short* eb      = (unsigned short*)(ws + 4194304);        // 32,768,000 B
    float*          esq     = (float*)(ws + 36962304);                //    128,000 B
    float*          cand_d2 = (float*)(ws + 37090304);                //     32,768 B
    int*            cand_ix = (int*)(ws + 37123072);                  //     32,768 B

    cvt_x_kernel<<<dim3(BLROWS * DDIM / 1024), dim3(256), 0, stream>>>(xt, xb);
    cvt_emb_kernel<<<dim3(VROWS / 8), dim3(256), 0, stream>>>(emb, eb, esq);
    gemm_argmin_kernel<<<dim3(BLROWS / ROWS_PB, VSLICES), dim3(256), 0, stream>>>(
        xb, eb, esq, cand_d2, cand_ix);
    combine_gather_kernel<<<dim3(BLROWS / 64), dim3(256), 0, stream>>>(
        cand_d2, cand_ix, emb, out);
}
